// LlamaAttention_90426241450007
// MI455X (gfx1250) — compile-verified, matrix operand order checked by reading, not run on hardware
//
#include <hip/hip_runtime.h>
#include <math.h>

// Problem constants (from reference): B=1, S=2048, HID=4096, NH=32, HD=128
static constexpr int SEQ   = 2048;
static constexpr int HIDN  = 4096;
static constexpr int NHEAD = 32;
static constexpr int HDIM  = 128;

typedef __attribute__((ext_vector_type(16))) _Float16 v16h;
typedef __attribute__((ext_vector_type(8)))  _Float16 v8h;
typedef __attribute__((ext_vector_type(8)))  float    v8f;

// ---------------------------------------------------------------------------
// CDNA5 WMMA: D(16x16 f32) = A(16x32 f16) * B(32x16 f16) + C
// ---------------------------------------------------------------------------
__device__ inline v8f wmma16(v16h a, v16h b, v8f c) {
  return __builtin_amdgcn_wmma_f32_16x16x32_f16(
      /*neg_a=*/false, a, /*neg_b=*/false, b,
      /*c_mod=*/(short)0, c, /*reuse_a=*/false, /*reuse_b=*/false);
}

// Per-lane fragment loaders. ISA 7.12.2 layout for 16-bit A (16x32 MxK):
//   lanes 0-15 : row M = lane,    K = {k0+0..7,  k0+16..23}
//   lanes 16-31: row M = lane-16, K = {k0+8..15, k0+24..31}
// Caller passes p = &X[row][k0 + ((lane&16)?8:0)]; we read p[0..7], p[16..23].
// B fragments use the mirrored layout with lanes indexing the 16 N columns,
// i.e. a lane holds a column of B == a row of B^T, so passing W (row-major
// [N][K]) rows directly yields B = W^T, which is exactly "X @ W.T".
__device__ inline v16h frag_from_h(const _Float16* __restrict__ p) {
  v8h lo = *(const v8h*)(p);
  v8h hi = *(const v8h*)(p + 16);
  v16h f;
#pragma unroll
  for (int i = 0; i < 8; ++i) { f[i] = lo[i]; f[i + 8] = hi[i]; }
  return f;
}

__device__ inline v16h frag_from_f32(const float* __restrict__ p) {
  v16h f;
#pragma unroll
  for (int i = 0; i < 8; ++i) f[i] = (_Float16)p[i];
#pragma unroll
  for (int i = 0; i < 8; ++i) f[i + 8] = (_Float16)p[16 + i];
  return f;
}

// ---------------------------------------------------------------------------
// C[M,N] = A[M,K](f32) @ W[N,K](f32)^T  -- inline f32->f16 convert on load,
// f32 accumulate in WMMA. Block = 128 thr (4 waves), 128x128 tile; wave 64x64.
// ---------------------------------------------------------------------------
__global__ __launch_bounds__(128) void gemm_awt_wmma(
    const float* __restrict__ A, const float* __restrict__ W,
    float* __restrict__ C, int M, int N, int K) {
  const int lane = threadIdx.x & 31;
  const int wave = threadIdx.x >> 5;
  const int M0 = blockIdx.x * 128 + (wave >> 1) * 64;
  const int N0 = blockIdx.y * 128 + (wave & 1) * 64;
  const int koff = (lane & 16) ? 8 : 0;
  const int rA = M0 + (lane & 15);
  const int rB = N0 + (lane & 15);

  v8f z = {};
  v8f acc[4][4];
#pragma unroll
  for (int i = 0; i < 4; ++i)
#pragma unroll
    for (int j = 0; j < 4; ++j) acc[i][j] = z;

  for (int k0 = 0; k0 < K; k0 += 32) {
    v16h a[4], b[4];
#pragma unroll
    for (int i = 0; i < 4; ++i)
      a[i] = frag_from_f32(A + (size_t)(rA + 16 * i) * K + k0 + koff);
#pragma unroll
    for (int j = 0; j < 4; ++j)
      b[j] = frag_from_f32(W + (size_t)(rB + 16 * j) * K + k0 + koff);
    if (k0 + 32 < K) {  // gfx1250 global_prefetch_b8 for the K-stream
      __builtin_prefetch(A + (size_t)rA * K + k0 + 32 + koff, 0, 0);
      __builtin_prefetch(W + (size_t)rB * K + k0 + 32 + koff, 0, 0);
    }
#pragma unroll
    for (int i = 0; i < 4; ++i)
#pragma unroll
      for (int j = 0; j < 4; ++j)
        acc[i][j] = wmma16(a[i], b[j], acc[i][j]);
  }

  // C/D layout: VGPR r -> M = r + 8*(lane>=16), N = lane&15 (ISA 7.12.2)
  const int rbase = M0 + ((lane & 16) ? 8 : 0);
  const int cbase = N0 + (lane & 15);
#pragma unroll
  for (int i = 0; i < 4; ++i)
#pragma unroll
    for (int j = 0; j < 4; ++j)
#pragma unroll
      for (int r = 0; r < 8; ++r)
        C[(size_t)(rbase + 16 * i + r) * N + cbase + 16 * j] = acc[i][j][r];
}

// ---------------------------------------------------------------------------
// RoPE on Q/K + repack into per-head f16 layouts:
//   QH/KH[h][s][d]  (rows contiguous for A and B^T fragments of QK^T)
//   VtH[h][d][s]    (V columns contiguous for B fragments of P*V)
// ---------------------------------------------------------------------------
__global__ __launch_bounds__(256) void rope_pack(
    const float* __restrict__ Qf, const float* __restrict__ Kf,
    const float* __restrict__ Vf, _Float16* __restrict__ QH,
    _Float16* __restrict__ KH, _Float16* __restrict__ VtH) {
  const int idx = blockIdx.x * 256 + threadIdx.x;  // exact grid: SEQ*HIDN/256
  const int d = idx & (HDIM - 1);
  const int h = (idx >> 7) & (NHEAD - 1);
  const int s = idx >> 12;
  const size_t src = (size_t)s * HIDN + h * HDIM + d;

  const int dr = d & 63;                                   // freq index
  const float inv = __expf(-(float)dr * (9.210340371976184f / 64.f)); // ln(1e4)
  const float th = (float)s * inv;                         // position_ids = s
  const float cs = cosf(th), sn = sinf(th);
  const int dp = (d < 64) ? d + 64 : d - 64;               // rotate_half pair
  const float sgn = (d < 64) ? -1.f : 1.f;
  const size_t srcp = (size_t)s * HIDN + h * HDIM + dp;

  const float q = Qf[src], k = Kf[src], v = Vf[src];
  const float qo = q * cs + sgn * Qf[srcp] * sn;
  const float ko = k * cs + sgn * Kf[srcp] * sn;

  QH[((size_t)h * SEQ + s) * HDIM + d] = (_Float16)qo;
  KH[((size_t)h * SEQ + s) * HDIM + d] = (_Float16)ko;
  VtH[((size_t)h * HDIM + d) * SEQ + s] = (_Float16)v;
}

// ---------------------------------------------------------------------------
// wave32 shuffle reductions
// ---------------------------------------------------------------------------
__device__ inline float wred_max(float v) {
#pragma unroll
  for (int o = 16; o > 0; o >>= 1) v = fmaxf(v, __shfl_xor(v, o, 32));
  return v;
}
__device__ inline float wred_sum(float v) {
#pragma unroll
  for (int o = 16; o > 0; o >>= 1) v += __shfl_xor(v, o, 32);
  return v;
}

// ---------------------------------------------------------------------------
// Async K-tile stager (gfx1250 GLOBAL_LOAD_ASYNC_TO_LDS_B128, ASYNCcnt).
// Moves one 16x128 f16 tile (4 KB, 16 rows x 256 B) into a bank-padded LDS
// tile (row stride 136 halves = 272 B, so the 16 lanes of a fragment read
// land in 16 distinct banks). Two rows per instruction (32 lanes x 16 B),
// 8 instructions per tile; completion tracked by ASYNCcnt.
// ---------------------------------------------------------------------------
static constexpr int KT_LD = 136;  // padded LDS row stride in halves

__device__ inline void stage_ktile_async(const _Float16* __restrict__ gsrc,
                                         const _Float16* lds_dst, int lane) {
  const int rsel = lane >> 4;          // which row of each 2-row pair
  const int cb = (lane & 15) * 16;     // byte offset within a 256 B row
#pragma unroll
  for (int q = 0; q < 8; ++q) {
    const int row = q * 2 + rsel;
    unsigned long long ga =
        (unsigned long long)(const char*)gsrc + (unsigned)(row * 256 + cb);
    unsigned la = (unsigned)(size_t)lds_dst + (unsigned)(row * (KT_LD * 2) + cb);
    asm volatile("global_load_async_to_lds_b128 %0, %1, off"
                 :: "v"(la), "v"(ga)
                 : "memory");
  }
}

// ---------------------------------------------------------------------------
// Fused attention: one wave per (head, 16-query-row block).
//   scores(16 x CE) via WMMA (K tiles async-double-buffered in LDS) -> LDS f32
//   causal mask + softmax + attention_dispense (row-local) in LDS
//   ctx(16x128) = P(16 x CE, f16 on-the-fly) x V via WMMA
// CE = round32((rb+1)*16): everything right of the causal frontier is exactly
// zero after softmax, so both phases stop there (~2x average work saved).
// Every strided column loop uses c = lane + 32*t, so each column is owned by
// one lane: no cross-lane LDS hazards inside softmax/dispense.
// ---------------------------------------------------------------------------
__global__ __launch_bounds__(32) void attn_fused(
    const _Float16* __restrict__ QH, const _Float16* __restrict__ KH,
    const _Float16* __restrict__ VtH, float* __restrict__ ctx) {
  __shared__ float sc[16][SEQ + 4];
  __shared__ _Float16 ktile[2][16 * KT_LD];
  const int lane = threadIdx.x & 31;
  const int h = blockIdx.x >> 7;           // 128 row-blocks per head
  const int rb = blockIdx.x & 127;
  const int S0 = rb * 16;
  const int ctn = (((rb + 1) * 16 + 31) >> 5) << 1;  // col tiles to compute
  const int CE = ctn * 16;                           // processed column extent
  const int koff = (lane & 16) ? 8 : 0;
  const int rl0 = (lane & 16) ? 8 : 0;
  const float scale = 0.08838834764831845f;  // 1/sqrt(128)
  v8f z = {};

  // Hoisted Q fragments (16 rows x 128 head-dim = 4 k-steps)
  const _Float16* qb = QH + ((size_t)h * SEQ + S0 + (lane & 15)) * HDIM;
  v16h qf[4];
#pragma unroll
  for (int kk = 0; kk < 4; ++kk) qf[kk] = frag_from_h(qb + kk * 32 + koff);

  // ---- scores = scale * Q K^T, causal-masked, into LDS ----
  const _Float16* khead = KH + (size_t)h * SEQ * HDIM;
  stage_ktile_async(khead, &ktile[0][0], lane);   // preload tile 0
  for (int ct = 0; ct < ctn; ++ct) {
    if (ct + 1 < ctn) {
      stage_ktile_async(khead + (size_t)(ct + 1) * 16 * HDIM,
                        &ktile[(ct + 1) & 1][0], lane);
      // newest 8 async ops may stay in flight; tile `ct` is complete
      asm volatile("s_wait_asynccnt 0x8" ::: "memory");
    } else {
      asm volatile("s_wait_asynccnt 0x0" ::: "memory");
    }
    const _Float16* kt = &ktile[ct & 1][(lane & 15) * KT_LD + koff];
    v8f acc = z;
#pragma unroll
    for (int kk = 0; kk < 4; ++kk)
      acc = wmma16(qf[kk], frag_from_h(kt + kk * 32), acc);
    const int col = ct * 16 + (lane & 15);
#pragma unroll
    for (int r = 0; r < 8; ++r) {
      const int rl = rl0 + r;
      sc[rl][col] = (col <= S0 + rl) ? acc[r] * scale : -3.402823466e38f;
    }
  }
  __syncthreads();

  // ---- softmax + attention_dispense per row (columns < CE only) ----
  for (int r = 0; r < 16; ++r) {
    float m = -3.402823466e38f;
    for (int c = lane; c < CE; c += 32) m = fmaxf(m, sc[r][c]);
    m = wred_max(m);
    float sum = 0.f;
    for (int c = lane; c < CE; c += 32) {
      float e = __expf(sc[r][c] - m);   // masked cols -> exp(-huge) = 0
      sc[r][c] = e;
      sum += e;
    }
    sum = wred_sum(sum);
    const float invs = 1.f / sum;
    for (int c = lane; c < CE; c += 32) sc[r][c] *= invs;

    const int rowg = S0 + r;
    if (rowg >= 512 && rowg < 1536) {
      const bool b1 = rowg < 1024;            // block (512,1024) vs (1024,1536)
      const int end = b1 ? 1024 : 1536;
      const float lb = 0.5f / (float)end;     // alpha/end
      // Columns in (rowg, end) are exact zeros: threshold is a no-op and they
      // add 0 to the elim sum, so bound both loops at eff = min(end, rowg+1).
      const int eff = (end < rowg + 1) ? end : rowg + 1;
      // threshold + halve column 0 (col 0 is always owned by lane 0)
      for (int c = lane; c < eff; c += 32) {
        float p = sc[r][c];
        p = (p >= lb) ? p : 0.f;
        if (c == 0) p *= 0.5f;
        sc[r][c] = p;
      }
      float s1 = 0.f;
      for (int c = lane; c < eff; c += 32) s1 += sc[r][c];
      const float elim = 1.f - wred_sum(s1);
      const int pw = b1 ? 512 : 1024;         // prev region width (<= rowg)
      const float tiny = b1 ? 0.001f : 0.01f; // soft-mask value
      float nz = 0.f;
      for (int c = lane; c < pw; c += 32) nz += (sc[r][c] >= lb) ? 1.f : tiny;
      nz = wred_sum(nz);
      const float add = elim / nz * 0.1f;     // beta = 0.1
      if (b1) {                                // start == interval branch: 10x
        for (int c = lane; c < 512; c += 32) {
          float pm = (sc[r][c] >= lb) ? 1.f : 0.001f;
          sc[r][c] += 10.f * add * pm;
        }
      } else {                                 // 3x to cols 0:512, 7x to 512:1024
        for (int c = lane; c < 512; c += 32) {
          float pm = (sc[r][c] >= lb) ? 1.f : 0.01f;
          sc[r][c] += 3.f * add * pm;
        }
        for (int c = 512 + lane; c < 1024; c += 32) {
          float pm = (sc[r][c] >= lb) ? 1.f : 0.01f;
          sc[r][c] += 7.f * add * pm;
        }
      }
    }
  }
  __syncthreads();

  // ---- ctx(16x128) = P(16 x CE) @ V(CE x 128), P read from LDS as f16 ----
  v8f o[8];
#pragma unroll
  for (int j = 0; j < 8; ++j) o[j] = z;
  for (int k0 = 0; k0 < CE; k0 += 32) {
    const float* pp = &sc[lane & 15][k0 + koff];
    v16h pa;
#pragma unroll
    for (int i = 0; i < 8; ++i) {
      pa[i] = (_Float16)pp[i];
      pa[i + 8] = (_Float16)pp[16 + i];
    }
#pragma unroll
    for (int j = 0; j < 8; ++j) {
      const _Float16* vb =
          VtH + ((size_t)h * HDIM + j * 16 + (lane & 15)) * SEQ + k0 + koff;
      o[j] = wmma16(pa, frag_from_h(vb), o[j]);
    }
  }
#pragma unroll
  for (int j = 0; j < 8; ++j)
#pragma unroll
    for (int r = 0; r < 8; ++r) {
      const int row = S0 + rl0 + r;
      const int d = j * 16 + (lane & 15);
      ctx[(size_t)row * HIDN + h * HDIM + d] = o[j][r];
    }
}

// ---------------------------------------------------------------------------
// Host-side orchestration (graph-capture safe: only kernel launches on stream)
// Inputs: 0=hidden_states 1=attention_mask 2=position_ids 3=wq 4=wk 5=wv 6=wo
// Workspace layout (176 MB): Qf|Kf|Vf (f32, 32MB ea) | QH|KH|VtH (f16, 16MB ea)
//                            | ctx (f32, 32MB)
// ---------------------------------------------------------------------------
extern "C" void kernel_launch(void* const* d_in, const int* in_sizes, int n_in,
                              void* d_out, int out_size, void* d_ws,
                              size_t ws_size, hipStream_t stream) {
  const float* hs = (const float*)d_in[0];
  const float* wq = (const float*)d_in[3];
  const float* wk = (const float*)d_in[4];
  const float* wv = (const float*)d_in[5];
  const float* wo = (const float*)d_in[6];
  float* out = (float*)d_out;

  const size_t E = (size_t)SEQ * HIDN;
  float* Qf = (float*)d_ws;
  float* Kf = Qf + E;
  float* Vf = Kf + E;
  _Float16* QH = (_Float16*)(Vf + E);
  _Float16* KH = QH + E;
  _Float16* VtH = KH + E;
  float* ctx = (float*)(VtH + E);

  dim3 gg(SEQ / 128, HIDN / 128);
  gemm_awt_wmma<<<gg, 128, 0, stream>>>(hs, wq, Qf, SEQ, HIDN, HIDN);
  gemm_awt_wmma<<<gg, 128, 0, stream>>>(hs, wk, Kf, SEQ, HIDN, HIDN);
  gemm_awt_wmma<<<gg, 128, 0, stream>>>(hs, wv, Vf, SEQ, HIDN, HIDN);
  rope_pack<<<(SEQ * HIDN) / 256, 256, 0, stream>>>(Qf, Kf, Vf, QH, KH, VtH);
  attn_fused<<<NHEAD * (SEQ / 16), 32, 0, stream>>>(QH, KH, VtH, ctx);
  gemm_awt_wmma<<<gg, 128, 0, stream>>>(ctx, wo, out, SEQ, HIDN, HIDN);
}
